// GCLSTM_83597243450106
// MI455X (gfx1250) — matrix operand order repaired and checked by reading, not compiled, not run on hardware
//
#include <hip/hip_runtime.h>

// ---------------------------------------------------------------------------
// GC-LSTM for MI455X (gfx1250, wave32, WMMA).
// B=64, H=168, N=512, F=8, P=24, NODE=0.
// All WMMA B-operands are stored transposed so fragment reads are contiguous
// 16-byte vector loads (global_load_b128 / ds_load_b128).
// ---------------------------------------------------------------------------

#define Bv   64
#define Hv   168
#define Nv   512
#define Fv   8
#define Pv   24
#define UNITS 128
#define GATES 512   // 4*UNITS

typedef __attribute__((ext_vector_type(16))) __bf16 v16bf;
typedef __attribute__((ext_vector_type(8)))  __bf16 v8bf;
typedef __attribute__((ext_vector_type(8)))  float  v8f;

// ---------------- WMMA helpers (layouts per CDNA5 ISA 7.12.2) ---------------

__device__ __forceinline__ v8f wmma_bf16(v16bf a, v16bf b, v8f c) {
  // (neg_a, A, neg_b, B, c_mod, C, reuse_a, reuse_b)
  return __builtin_amdgcn_wmma_f32_16x16x32_bf16(false, a, false, b, (short)0, c,
                                                 false, false);
}

// A fragment: 16x32 bf16 from row-major source (ld multiple of 8, 16B-aligned
// rows). lane L<16: row M=L, K groups {0..7,16..23}; lane>=16: {8..15,24..31}.
__device__ __forceinline__ v16bf frag_a_bf16(const __bf16* base, int ld, int k0,
                                             int lane) {
  const __bf16* rp = base + (lane & 15) * ld + k0 + ((lane >> 4) << 3);
  v8bf lo = *(const v8bf*)(rp);
  v8bf hi = *(const v8bf*)(rp + 16);
  return __builtin_shufflevector(lo, hi, 0, 1, 2, 3, 4, 5, 6, 7,
                                 8, 9, 10, 11, 12, 13, 14, 15);
}

// B fragment: 32x16 bf16 read from a TRANSPOSED source BT (N rows x K cols,
// leading dim ldT, multiple of 8). lane L: col N = L&15 -> one BT row;
// K groups {0..7,16..23} (L<16) / {8..15,24..31} (L>=16) -> two 16-byte loads.
__device__ __forceinline__ v16bf frag_bT_bf16(const __bf16* baseT, int ldT,
                                              int k0, int col0, int lane) {
  const __bf16* p = baseT + (col0 + (lane & 15)) * ldT + k0 + ((lane >> 4) << 3);
  v8bf lo = *(const v8bf*)(p);
  v8bf hi = *(const v8bf*)(p + 16);
  return __builtin_shufflevector(lo, hi, 0, 1, 2, 3, 4, 5, 6, 7,
                                 8, 9, 10, 11, 12, 13, 14, 15);
}

__device__ __forceinline__ float sigm(float x) { return 1.f / (1.f + __expf(-x)); }
// tanh via the native v_exp_f32 path (avoids the heavyweight OCML expansion)
__device__ __forceinline__ float fast_tanh(float x) {
  return 1.f - 2.f / (__expf(2.f * x) + 1.f);
}

// ---------------- 1) node summaries: (B,H,N,F) -> nf (B,N,7) ----------------
__global__ void node_summary_kernel(const float* __restrict__ inp,
                                    float* __restrict__ nf) {
  int idx = blockIdx.x * blockDim.x + threadIdx.x;
  if (idx >= Bv * Nv) return;
  int b = idx >> 9, n = idx & (Nv - 1);
  const float* p = inp + (size_t)b * Hv * Nv * Fv + (size_t)n * Fv;
  float s1 = 0, s2 = 0, s3 = 0, s4 = 0, s1h = 0, s2h = 0, st = 0;
  for (int t = 0; t < Hv; ++t) {
    float x = p[(size_t)t * Nv * Fv];
    float x2 = x * x;
    s1 += x; s2 += x2; s3 += x2 * x; s4 += x2 * x2;
    if (t >= Hv / 2) { s1h += x; s2h += x2; }
    st += ((float)t - 83.5f) * x;
  }
  const float invH = 1.f / (float)Hv, invHh = 2.f / (float)Hv;
  float mu  = s1 * invH;
  float m2  = s2 * invH - mu * mu;
  float m3  = s3 * invH - 3.f * mu * s2 * invH + 2.f * mu * mu * mu;
  float m4  = s4 * invH - 4.f * mu * s3 * invH + 6.f * mu * mu * s2 * invH
              - 3.f * mu * mu * mu * mu;
  float muh = s1h * invHh;
  float stdf = sqrtf(fmaxf(m2, 0.f));
  float stdh = sqrtf(fmaxf(s2h * invHh - muh * muh, 0.f));
  float skew = 0.f, kurt = 0.f;
  if (m2 > 1e-20f) {
    skew = m3 / (stdf * stdf * stdf);
    kurt = m4 / (m2 * m2) - 3.f;
  }
  // sum_t (t-83.5)^2 = H*(H^2-1)/12 = 395122
  float slope = st / 395122.0f;
  float* o = nf + (size_t)idx * 7;
  o[0] = mu; o[1] = muh; o[2] = stdf; o[3] = stdh;
  o[4] = skew; o[5] = kurt; o[6] = slope;
}

// ---------------- 2) small conversions / projections ------------------------
__global__ void f32_to_bf16_kernel(const float* __restrict__ src,
                                   __bf16* __restrict__ dst, int n) {
  int i = blockIdx.x * blockDim.x + threadIdx.x;
  if (i < n) dst[i] = (__bf16)src[i];
}

// X1T[b,c,n] = sum_k nf[b,n,k] * w_gcn1[k,c]   (7 -> 32), bf16, TRANSPOSED out
__global__ void proj1_kernel(const float* __restrict__ nf,
                             const float* __restrict__ wg,
                             __bf16* __restrict__ X1T) {
  int idx = blockIdx.x * blockDim.x + threadIdx.x;  // ((b*32+c)*512+n)
  if (idx >= Bv * 32 * Nv) return;
  int n = idx & (Nv - 1), c = (idx >> 9) & 31, b = idx >> 14;
  const float* r = nf + ((size_t)b * Nv + n) * 7;
  float s = 0.f;
#pragma unroll
  for (int k = 0; k < 7; ++k) s += r[k] * wg[k * 32 + c];
  X1T[idx] = (__bf16)s;
}

// X2T[b,c,n] = sum_k H1T[b,k,n] * w_gcn2[k,c]  (32 -> 16), bf16, TRANSPOSED
__global__ void proj2_kernel(const __bf16* __restrict__ H1T,
                             const float* __restrict__ wg,
                             __bf16* __restrict__ X2T) {
  int idx = blockIdx.x * blockDim.x + threadIdx.x;  // ((b*16+c)*512+n)
  if (idx >= Bv * 16 * Nv) return;
  int n = idx & (Nv - 1), c = (idx >> 9) & 15, b = idx >> 13;
  const __bf16* hb = H1T + (size_t)b * 32 * Nv + n;
  float s = 0.f;
#pragma unroll
  for (int k = 0; k < 32; ++k) s += (float)hb[(size_t)k * Nv] * wg[k * 16 + c];
  X2T[idx] = (__bf16)s;
}

// ---------------- 3) GCN propagation: OutT[b] = relu(adj @ X[b] + bias)^T ---
// grid = (Mtiles=32, 1, B), block = 32 (one wave). The wave computes all C/16
// column tiles for its row tile, reusing each adj A fragment across them.
template <int C>
__global__ void gcn_wmma_kernel(const __bf16* __restrict__ adjb,
                                const __bf16* __restrict__ XT,
                                const float* __restrict__ bias,
                                __bf16* __restrict__ OutT) {
  const int lane = threadIdx.x;
  const int mtile = blockIdx.x, b = blockIdx.z;
  const __bf16* Xb = XT + (size_t)b * C * Nv;
  constexpr int CT = C / 16;
  v8f acc[CT];
#pragma unroll
  for (int c = 0; c < CT; ++c) acc[c] = (v8f){};
#pragma unroll 4
  for (int kc = 0; kc < Nv / 32; ++kc) {
    v16bf a = frag_a_bf16(adjb + (size_t)mtile * 16 * Nv, Nv, kc * 32, lane);
#pragma unroll
    for (int c = 0; c < CT; ++c) {
      v16bf bb = frag_bT_bf16(Xb, Nv, kc * 32, c * 16, lane);
      acc[c] = wmma_bf16(a, bb, acc[c]);
    }
  }
  const int row0 = mtile * 16 + ((lane >> 4) << 3);
#pragma unroll
  for (int c = 0; c < CT; ++c) {
    const int col = c * 16 + (lane & 15);
    const float bv = bias[col];
    v8bf pack;
#pragma unroll
    for (int v = 0; v < 8; ++v) {
      float r = acc[c][v] + bv;
      pack[v] = (__bf16)(r < 0.f ? 0.f : r);
    }
    // D rows v..v+7 for this lane are consecutive node indices in OutT
    *(v8bf*)&OutT[((size_t)b * C + col) * Nv + row0] = pack;
  }
}

// ---------------- 4) conv1 -> pool -> conv2 -> gcn_out (feat[:,0:64]) -------
// one block per batch, 64 threads; GT is (b, 16, 512) transposed layout
__global__ void conv_head_kernel(const __bf16* __restrict__ GT,
                                 const float* __restrict__ w1,
                                 const float* __restrict__ b1,
                                 const float* __restrict__ w2,
                                 const float* __restrict__ b2,
                                 float* __restrict__ feat) {
  __shared__ float c1s[4][16];
  __shared__ float ps[4][8];
  const int b = blockIdx.x, tid = threadIdx.x;
  {
    int o = tid >> 4, s = tid & 15;
    float acc = b1[o];
#pragma unroll
    for (int dk = 0; dk < 3; ++dk) {
      int sp = s + dk - 1;
      if (sp < 0 || sp >= 16) continue;
      const __bf16* gr = GT + ((size_t)b * 16 + sp) * Nv;   // contiguous in i
      const float* wr = w1 + (size_t)dk * Nv * 4 + o;
      for (int i = 0; i < Nv; ++i) acc += (float)gr[i] * wr[(size_t)i * 4];
    }
    c1s[o][s] = acc;
  }
  __syncthreads();
  if (tid < 32) {
    int o = tid >> 3, s = tid & 7;
    ps[o][s] = 0.5f * (c1s[o][2 * s] + c1s[o][2 * s + 1]);
  }
  __syncthreads();
  if (tid < 32) {
    int o = tid >> 3, s = tid & 7;
    float acc = b2[o];
#pragma unroll
    for (int dk = 0; dk < 3; ++dk) {
      int sp = s + dk - 1;
      if (sp >= 0 && sp < 8)
#pragma unroll
        for (int i = 0; i < 4; ++i) acc += ps[i][sp] * w2[dk * 16 + i * 4 + o];
    }
    feat[(size_t)b * 192 + o * 8 + s]      = acc;       // c2 flat
    feat[(size_t)b * 192 + 32 + o * 8 + s] = ps[o][s];  // p flat
  }
}

// ---------------- 5) LSTM with persistent-c WMMA recurrence -----------------
// grid.x = 4 batch tiles (16 rows each); block = 256 (8 waves).
// Wave w owns unit-tile w (16 of 128 units): computes z for gates i/f/g/o at
// columns {0,128,256,384}+16w via v_wmma_f32_16x16x32_bf16; the cell state c
// lives in a persistent 8xf32 accumulator fragment across all T steps.
// LDS: WT = stacked weights TRANSPOSED (512 x UP bf16, UP = KTOT+8 pad) so B
// fragments are two ds_load_b128 per gate; U = [x_t;h] (16 x UP bf16).
// K-chunk loop is fully unrolled and double-buffered: chunk kc+1's 10
// ds_load_b128 are issued before chunk kc's 4 WMMAs so DS waits are partial.
template <int KX, int MODE>
__global__ void lstm_wmma_kernel(const float* __restrict__ xf,   // mode 1 input
                                 const __bf16* __restrict__ xb,  // mode 2 input
                                 const float* __restrict__ k,
                                 const float* __restrict__ rk,
                                 const float* __restrict__ bias,
                                 __bf16* __restrict__ seq_out,
                                 float* __restrict__ fin_out) {
  constexpr int KTOT = KX + UNITS;
  constexpr int UP   = KTOT + 8;        // padded leading dim (multiple of 8)
  constexpr int NCH  = KTOT / 32;
  extern __shared__ __align__(16) char smem_raw[];
  __bf16* WT = (__bf16*)smem_raw;       // [GATES][UP]  (transposed weights)
  __bf16* U  = WT + (size_t)GATES * UP; // [16][UP]

  const int tid  = threadIdx.x;
  const int lane = tid & 31;
  const int wv   = tid >> 5;            // unit tile 0..7
  const int mtile = blockIdx.x;         // batch tile 0..3

  // Stage stacked weights [k (Fv rows for mode1, zero-padded to KX); rk],
  // transposed into LDS.
  for (int idx = tid; idx < KTOT * GATES; idx += 256) {
    int r = idx >> 9, c = idx & (GATES - 1);   // coalesced global reads
    float v = 0.f;
    if (r < KX) { if (MODE != 1 || r < Fv) v = k[r * GATES + c]; }
    else          v = rk[(r - KX) * GATES + c];
    WT[(size_t)c * UP + r] = (__bf16)v;
  }
  for (int c = tid; c < GATES; c += 256)       // zero the pad columns
#pragma unroll
    for (int r = KTOT; r < UP; ++r) WT[(size_t)c * UP + r] = (__bf16)0.f;
  for (int idx = tid; idx < 16 * UP; idx += 256) U[idx] = (__bf16)0.f;
  __syncthreads();

  v8f cst = {};
  const int ncol = lane & 15;
  const int unit = wv * 16 + ncol;
  const float bi = bias[          unit];
  const float bf = bias[UNITS   + unit];
  const float bg = bias[2*UNITS + unit];
  const float bo = bias[3*UNITS + unit];

  for (int t = 0; t < Hv; ++t) {
    // stage x_t for this block's 16 batch rows (vectorized)
    if (MODE == 1) {
      if (tid < 32) {
        int bl = tid >> 1, f4 = (tid & 1) * 4;
        int b = mtile * 16 + bl;
        const float4 xv =
            *(const float4*)&xf[((size_t)b * Hv + t) * Nv * Fv + f4];
        __bf16* up = &U[bl * UP + f4];
        up[0] = (__bf16)xv.x; up[1] = (__bf16)xv.y;
        up[2] = (__bf16)xv.z; up[3] = (__bf16)xv.w;
      }
    } else {
      int bl = tid >> 4, f8 = (tid & 15) * 8;   // 256 threads = 16x16 v8bf
      int b = mtile * 16 + bl;
      *(v8bf*)&U[bl * UP + f8] =
          *(const v8bf*)&xb[((size_t)b * Hv + t) * UNITS + f8];
    }
    __syncthreads();

    v8f zi = {}, zf = {}, zg = {}, zo = {};
    // prologue: chunk 0 fragments
    v16bf a  = frag_a_bf16(U, UP, 0, lane);
    v16bf b0 = frag_bT_bf16(WT, UP, 0,           wv * 16, lane);
    v16bf b1 = frag_bT_bf16(WT, UP, 0, UNITS   + wv * 16, lane);
    v16bf b2 = frag_bT_bf16(WT, UP, 0, 2*UNITS + wv * 16, lane);
    v16bf b3 = frag_bT_bf16(WT, UP, 0, 3*UNITS + wv * 16, lane);
#pragma unroll
    for (int kc = 0; kc < NCH; ++kc) {
      v16bf an = a, c0 = b0, c1 = b1, c2 = b2, c3 = b3;
      if (kc + 1 < NCH) {   // prefetch next chunk before consuming this one
        const int kn = (kc + 1) * 32;
        an = frag_a_bf16(U, UP, kn, lane);
        c0 = frag_bT_bf16(WT, UP, kn,           wv * 16, lane);
        c1 = frag_bT_bf16(WT, UP, kn, UNITS   + wv * 16, lane);
        c2 = frag_bT_bf16(WT, UP, kn, 2*UNITS + wv * 16, lane);
        c3 = frag_bT_bf16(WT, UP, kn, 3*UNITS + wv * 16, lane);
      }
      zi = wmma_bf16(a, b0, zi);
      zf = wmma_bf16(a, b1, zf);
      zg = wmma_bf16(a, b2, zg);
      zo = wmma_bf16(a, b3, zo);
      a = an; b0 = c0; b1 = c1; b2 = c2; b3 = c3;
    }

    float hv[8];
#pragma unroll
    for (int v = 0; v < 8; ++v) {
      float iv = sigm(zi[v] + bi);
      float fv = sigm(zf[v] + bf);
      float gv = fast_tanh(zg[v] + bg);
      float ov = sigm(zo[v] + bo);
      float cv = fv * cst[v] + iv * gv;
      cst[v] = cv;
      hv[v] = ov * fast_tanh(cv);
    }
    __syncthreads();  // all U reads of this step complete before h writeback
#pragma unroll
    for (int v = 0; v < 8; ++v) {
      int bl = v + ((lane >> 4) << 3);  // D-fragment row
      U[bl * UP + KX + unit] = (__bf16)hv[v];
      int b = mtile * 16 + bl;
      if (seq_out)
        seq_out[((size_t)b * Hv + t) * UNITS + unit] = (__bf16)hv[v];
      if (fin_out && t == Hv - 1)
        fin_out[(size_t)b * 192 + 64 + unit] = hv[v];
    }
    // next-iteration x staging writes a disjoint LDS region; the barrier at
    // the top of the loop orders all writes before the next fragment reads.
  }
}

// ---------------- 6) output head: out = feat @ w_out + b_out ----------------
__global__ void head_kernel(const float* __restrict__ feat,
                            const float* __restrict__ w,
                            const float* __restrict__ bo,
                            float* __restrict__ out) {
  int idx = blockIdx.x * blockDim.x + threadIdx.x;
  if (idx >= Bv * Pv) return;
  int b = idx / Pv, p = idx - b * Pv;
  float s = bo[p];
  const float* f = feat + (size_t)b * 192;
  for (int c = 0; c < 192; ++c) s += f[c] * w[c * Pv + p];
  out[idx] = s;
}

// ---------------------------------------------------------------------------
extern "C" void kernel_launch(void* const* d_in, const int* in_sizes, int n_in,
                              void* d_out, int out_size, void* d_ws,
                              size_t ws_size, hipStream_t stream) {
  const float* inputs  = (const float*)d_in[0];
  const float* adj     = (const float*)d_in[1];
  const float* w_gcn1  = (const float*)d_in[2];
  const float* b_gcn1  = (const float*)d_in[3];
  const float* w_gcn2  = (const float*)d_in[4];
  const float* b_gcn2  = (const float*)d_in[5];
  const float* w_conv1 = (const float*)d_in[6];
  const float* b_conv1 = (const float*)d_in[7];
  const float* w_conv2 = (const float*)d_in[8];
  const float* b_conv2 = (const float*)d_in[9];
  const float* k_l1    = (const float*)d_in[10];
  const float* rk_l1   = (const float*)d_in[11];
  const float* b_l1    = (const float*)d_in[12];
  const float* k_l2    = (const float*)d_in[13];
  const float* rk_l2   = (const float*)d_in[14];
  const float* b_l2    = (const float*)d_in[15];
  const float* w_out   = (const float*)d_in[16];
  const float* b_out   = (const float*)d_in[17];
  float* out = (float*)d_out;

  // workspace carve-up (~10.5 MB total)
  char* w = (char*)d_ws;
  auto take = [&](size_t bytes) {
    char* p = w;
    w += (bytes + 255) & ~(size_t)255;
    return p;
  };
  float*  nf   = (float*)take((size_t)Bv * Nv * 7 * 4);
  __bf16* adjb = (__bf16*)take((size_t)Nv * Nv * 2);
  __bf16* X1T  = (__bf16*)take((size_t)Bv * 32 * Nv * 2);
  __bf16* H1T  = (__bf16*)take((size_t)Bv * 32 * Nv * 2);
  __bf16* X2T  = (__bf16*)take((size_t)Bv * 16 * Nv * 2);
  __bf16* GT   = (__bf16*)take((size_t)Bv * 16 * Nv * 2);
  __bf16* l1   = (__bf16*)take((size_t)Bv * Hv * UNITS * 2);
  float*  feat = (float*)take((size_t)Bv * 192 * 4);

  // dynamic-LDS sizes: (GATES + 16) rows x UP bf16
  const int up1 = (32 + UNITS) + 8, up2 = (128 + UNITS) + 8;
  const int shmem1 = (GATES + 16) * up1 * 2;  // ~177 KB
  const int shmem2 = (GATES + 16) * up2 * 2;  // ~279 KB
  auto* lstm1 = lstm_wmma_kernel<32, 1>;
  auto* lstm2 = lstm_wmma_kernel<128, 2>;
  hipFuncSetAttribute((const void*)lstm1,
                      hipFuncAttributeMaxDynamicSharedMemorySize, shmem1);
  hipFuncSetAttribute((const void*)lstm2,
                      hipFuncAttributeMaxDynamicSharedMemorySize, shmem2);

  // 1) node summaries
  node_summary_kernel<<<(Bv * Nv + 255) / 256, 256, 0, stream>>>(inputs, nf);
  // 2) adj -> bf16
  f32_to_bf16_kernel<<<(Nv * Nv + 255) / 256, 256, 0, stream>>>(adj, adjb,
                                                                Nv * Nv);
  // 3) GCN layer 1
  proj1_kernel<<<(Bv * 32 * Nv + 255) / 256, 256, 0, stream>>>(nf, w_gcn1, X1T);
  gcn_wmma_kernel<32><<<dim3(Nv / 16, 1, Bv), 32, 0, stream>>>(adjb, X1T,
                                                               b_gcn1, H1T);
  // 4) GCN layer 2
  proj2_kernel<<<(Bv * 16 * Nv + 255) / 256, 256, 0, stream>>>(H1T, w_gcn2, X2T);
  gcn_wmma_kernel<16><<<dim3(Nv / 16, 1, Bv), 32, 0, stream>>>(adjb, X2T,
                                                               b_gcn2, GT);
  // 5) conv head -> feat[:, 0:64]
  conv_head_kernel<<<Bv, 64, 0, stream>>>(GT, w_conv1, b_conv1, w_conv2,
                                          b_conv2, feat);
  // 6) LSTM1 (x: inputs[:, :, 0, :], KX=32 pad of 8 real) -> l1 sequence
  lstm1<<<4, 256, shmem1, stream>>>(inputs, (const __bf16*)nullptr, k_l1,
                                    rk_l1, b_l1, l1, nullptr);
  // 7) LSTM2 (x: l1, KX=128) -> feat[:, 64:192]
  lstm2<<<4, 256, shmem2, stream>>>(nullptr, l1, k_l2, rk_l2, b_l2, nullptr,
                                    feat);
  // 8) output head
  head_kernel<<<(Bv * Pv + 255) / 256, 256, 0, stream>>>(feat, w_out, b_out,
                                                         out);
  (void)in_sizes; (void)n_in; (void)out_size; (void)ws_size;
}
